// MyEncoder_86088324481940
// MI455X (gfx1250) — compile-verified, matrix operand order checked by reading, not run on hardware
//
#include <hip/hip_runtime.h>
#include <hip/hip_bf16.h>
#include <math.h>

typedef __attribute__((ext_vector_type(16))) _Float16 v16h;
typedef __attribute__((ext_vector_type(8)))  _Float16 v8h;
typedef __attribute__((ext_vector_type(8)))  float    v8f;

#define BQ  8
#define SEQ 4096
#define EMB 128
#define DH  24
#define DP  32    // head dim padded to WMMA K
#define KT  64    // keys staged per LDS tile
#define NT  (SEQ / KT)

#define LDK 40    // LDS ld (halves) for K tile rows of 32  (+8 pad)
#define LDV 72    // LDS ld (halves) for Vt tile rows of 64 (+8 pad)
#define LDP 40    // LDS ld for P tile rows of 32
#define LDX 136   // LDS ld for x tile rows of 128 (+8 pad)

#if defined(__has_builtin)
#if __has_builtin(__builtin_amdgcn_tensor_load_to_lds) && __has_builtin(__builtin_amdgcn_s_wait_tensorcnt)
#define HAVE_TDM 1
#endif
#endif
#ifndef HAVE_TDM
#define HAVE_TDM 0
#endif

// ---------------------------------------------------------------------------
// Fragment loads. Per CDNA5 ISA 7.12.2, a 16-bit A fragment per lane is two
// CONTIGUOUS 8-half runs: k in [hi,hi+8) and [16+hi,16+hi+8), hi = (lane&16)?8:0.
// So a row-major source needs exactly two b128 loads per lane.
// B fragment of W[k][n] == A fragment of W^T[n][k], so all B sources are stored
// transposed and use the same wide loader. ld must be a multiple of 8 halves.
// ---------------------------------------------------------------------------
__device__ __forceinline__ v16h load_frag_rowmajor(const _Float16* base, int ld, int lane) {
    const _Float16* r = base + (size_t)(lane & 15) * ld + ((lane & 16) ? 8 : 0);
    v8h lo = *(const v8h*)(r);
    v8h hi = *(const v8h*)(r + 16);
    return __builtin_shufflevector(lo, hi, 0, 1, 2, 3, 4, 5, 6, 7,
                                           8, 9, 10, 11, 12, 13, 14, 15);
}

__device__ __forceinline__ v8f wmma_f16(v16h a, v16h b, v8f c) {
    return __builtin_amdgcn_wmma_f32_16x16x32_f16(false, a, false, b, (short)0, c, false, false);
}

#if HAVE_TDM
typedef __attribute__((ext_vector_type(4))) unsigned int v4u;
typedef __attribute__((ext_vector_type(8))) int v8i;
typedef __attribute__((ext_vector_type(4))) int v4i;

union TdmG0 { unsigned long long q[2]; v4u v; };
union TdmG1 { unsigned long long q[4]; v8i v; };

// Build a D# (ISA 8.3/8.4) and issue TENSOR_LOAD_TO_LDS. 2-byte elements.
// pad_amount_dw != 0 enables LDS padding: pad_interval_code per ISA (3=16dw, 4=32dw).
// 6-arg builtin: (uint32x4 g0, int32x8 g1, int32x4, int32x4, int32x8, i32 cpol)
__device__ __forceinline__ void tdm_load_2d(
    unsigned lds_addr, unsigned long long gaddr,
    unsigned tile0, unsigned tile1,
    unsigned long long tensor0, unsigned long long tensor1,
    unsigned long long stride0,
    unsigned pad_interval_code, unsigned pad_amount_dw)
{
    TdmG0 g0;
    g0.q[0] = 1ull | ((unsigned long long)lds_addr << 32);            // count=1 | lds_addr
    g0.q[1] = (gaddr & ((1ull << 57) - 1)) | (2ull << 62);            // global_addr | type=2
    TdmG1 g1;
    unsigned flags = (1u << 16);                                      // data_size = 1 (2B)
    if (pad_amount_dw)
        flags |= (1u << 20) | (pad_interval_code << 22) | ((pad_amount_dw - 1) << 25);
    g1.q[0] = (unsigned long long)flags | ((tensor0 & 0xFFFFull) << 48);
    g1.q[1] = ((tensor0 >> 16) & 0xFFFFull)
            | ((tensor1 & 0xFFFFFFFFull) << 16)
            | ((unsigned long long)(tile0 & 0xFFFFu) << 48);
    g1.q[2] = (unsigned long long)(tile1 & 0xFFFFu) | ((stride0 & 0xFFFFFFFFull) << 32);
    g1.q[3] = (stride0 >> 32) & 0xFFFFull;                            // dim1_stride unused (2D)
    v4i z4 = {};
    v8i z8 = {};
    __builtin_amdgcn_tensor_load_to_lds(g0.v, g1.v, z4, z4, z8, 0);
}
#endif

// ---------------------------------------------------------------------------
// Kernel 1: Q/K/V projections. Block = 256 threads (8 waves), 64 tokens.
// Q pre-scaled by log2(e)/sqrt(EMB) (softmax runs in exp2 domain).
// V written TRANSPOSED: Vt[b][d][s]. Per-matrix code specialized at compile
// time so epilogues are straight-line stores (no exec-mask select bloat).
// ---------------------------------------------------------------------------
#define QSCALE 0.12751864f   // (1/sqrt(128)) * log2(e)

template <int M>
__device__ __forceinline__ void proj_tile(
    int t, int lane, const _Float16* ldsX, const _Float16* ldsWtM,
    const float* ldsBM, long tokBase,
    _Float16* __restrict__ Qh, _Float16* __restrict__ Kh, _Float16* __restrict__ Vt)
{
    v8f c0 = {}; v8f c1 = {};
#pragma unroll
    for (int kc = 0; kc < 4; ++kc) {
        v16h a  = load_frag_rowmajor(ldsX + t * 16 * LDX + kc * 32, LDX, lane);
        v16h b0 = load_frag_rowmajor(ldsWtM + kc * 32,            LDX, lane);
        v16h b1 = load_frag_rowmajor(ldsWtM + 16 * LDX + kc * 32, LDX, lane);
        c0 = wmma_f16(a, b0, c0);
        c1 = wmma_f16(a, b1, c1);
    }
    const int colL = lane & 15;
    const int rhi  = (lane & 16) ? 8 : 0;
#pragma unroll
    for (int r = 0; r < 8; ++r) {
        long tok = tokBase + t * 16 + r + rhi;
        float v0 = c0[r] + ldsBM[colL];
        float v1 = c1[r] + ldsBM[16 + colL];
        if (M == 0) { v0 *= QSCALE; v1 *= QSCALE; }
        if (M == 2) {            // V transposed: Vt[(b*DP + d)*SEQ + s]
            long b = tok >> 12, s = tok & (SEQ - 1);
            Vt[(b * DP + colL)      * SEQ + s] = (_Float16)v0;
            Vt[(b * DP + 16 + colL) * SEQ + s] = (_Float16)v1;
        } else {
            _Float16* dst = (M == 0) ? Qh : Kh;
            dst[tok * DP + colL]      = (_Float16)v0;
            dst[tok * DP + 16 + colL] = (_Float16)v1;
        }
    }
}

__global__ __launch_bounds__(256) void qkv_proj_kernel(
    const float* __restrict__ x,
    const float* __restrict__ wQ, const float* __restrict__ bQ,
    const float* __restrict__ wK, const float* __restrict__ bK,
    const float* __restrict__ wV, const float* __restrict__ bV,
    _Float16* __restrict__ Qh, _Float16* __restrict__ Kh, _Float16* __restrict__ Vt)
{
    __shared__ __align__(16) _Float16 ldsX[64 * LDX];          // ~17.4 KB
    __shared__ __align__(16) _Float16 ldsWt[3][DP * LDX];      // transposed [n][k], 26 KB
    __shared__ float ldsB[3][DP];

    const int tid  = threadIdx.x;
    const int lane = tid & 31;
    const int wv   = tid >> 5;
    const long tokBase = (long)blockIdx.x * 64;

    for (int i = tid; i < 64 * EMB; i += 256) {
        int r = i >> 7, c = i & 127;
        ldsX[r * LDX + c] = (_Float16)x[tokBase * EMB + i];
    }
    const float* wsrc[3] = { wQ, wK, wV };
    const float* bsrc[3] = { bQ, bK, bV };
    for (int m = 0; m < 3; ++m) {
        for (int i = tid; i < DP * EMB; i += 256) {
            int c = i >> 7, k = i & 127;                       // c = out col, k = emb
            ldsWt[m][c * LDX + k] = (c < DH) ? (_Float16)wsrc[m][k * DH + c] : (_Float16)0.0f;
        }
        if (tid < DP) ldsB[m][tid] = (tid < DH) ? bsrc[m][tid] : 0.0f;
    }
    __syncthreads();

    if (wv < 4) {
        proj_tile<0>(wv, lane, ldsX, ldsWt[0], ldsB[0], tokBase, Qh, Kh, Vt);
        proj_tile<2>(wv, lane, ldsX, ldsWt[2], ldsB[2], tokBase, Qh, Kh, Vt);
    } else {
        proj_tile<1>(wv - 4, lane, ldsX, ldsWt[1], ldsB[1], tokBase, Qh, Kh, Vt);
    }
}

// ---------------------------------------------------------------------------
// Kernel 2: fused flash attention. Block = 8 waves = 128 queries, one batch.
// K/V tiles (64 keys) staged to double-buffered LDS by the Tensor Data Mover
// (fallback: cooperative loads), overlapped with compute. 8 WMMAs per tile.
// Softmax runs in exp2 domain (scale folded into Q), so v_exp_f32 direct.
// ---------------------------------------------------------------------------
__global__ __launch_bounds__(256) void attn_kernel(
    const _Float16* __restrict__ Qh, const _Float16* __restrict__ Kh,
    const _Float16* __restrict__ Vt, _Float16* __restrict__ Hh)
{
    __shared__ __align__(16) _Float16 ldsK[2][KT * LDK];       // [key][d], padded
    __shared__ __align__(16) _Float16 ldsV[2][DP * LDV];       // [d][key], padded
    __shared__ __align__(16) _Float16 ldsP[8][16 * LDP];       // per-wave P relayout

    const int tid  = threadIdx.x;
    const int lane = tid & 31;
    const int wv   = tid >> 5;
    const int b    = blockIdx.x >> 5;                          // 32 blocks per batch
    const long qTok = (long)b * SEQ + (long)(blockIdx.x & 31) * 128 + wv * 16;
    const long kvTok = (long)b * SEQ;

    v16h qf = load_frag_rowmajor(Qh + qTok * DP, DP, lane);    // 16q x 32d, one K-step

    v8f o0 = {}; v8f o1 = {};
    float mrun[8], lrun[8];
#pragma unroll
    for (int r = 0; r < 8; ++r) { mrun[r] = -INFINITY; lrun[r] = 0.0f; }

    _Float16* P = &ldsP[wv][0];
    const int colL = lane & 15;
    const int rhi  = (lane & 16) ? 8 : 0;

    auto stage = [&](int buf, int tile) {
        const long kb = (long)tile * KT;
#if HAVE_TDM
        if (wv == 0) {
            // K tile: 64 rows x 32 halves contiguous; pad 4 dwords every 16 dwords -> ld 40.
            tdm_load_2d((unsigned)(unsigned long long)(&ldsK[buf][0]),
                        (unsigned long long)(const void*)(Kh + (kvTok + kb) * DP),
                        KT * DP, 0, KT * DP, 1, 0, /*interval 16dw*/3, /*pad*/4);
            // Vt tile: 2D, 32 rows (d) x 64 keys, row stride SEQ; pad 4 dwords / 32 dwords -> ld 72.
            tdm_load_2d((unsigned)(unsigned long long)(&ldsV[buf][0]),
                        (unsigned long long)(const void*)(Vt + (long)b * DP * SEQ + kb),
                        KT, DP, SEQ, DP, SEQ, /*interval 32dw*/4, /*pad*/4);
        }
#else
        const unsigned* ks = (const unsigned*)(Kh + (kvTok + kb) * DP);
        unsigned* kd = (unsigned*)&ldsK[buf][0];
        for (int i = tid; i < KT * DP / 2; i += 256) {         // u32 chunks
            int r = i >> 4, c = i & 15;
            kd[r * (LDK / 2) + c] = ks[i];
        }
        unsigned* vd = (unsigned*)&ldsV[buf][0];
        for (int i = tid; i < DP * KT / 2; i += 256) {
            int d = i >> 5, c = i & 31;
            const unsigned* vs = (const unsigned*)(Vt + ((long)b * DP + d) * SEQ + kb);
            vd[d * (LDV / 2) + c] = vs[c];
        }
#endif
    };

    stage(0, 0);
    for (int kt = 0; kt < NT; ++kt) {
        const int cur = kt & 1;
#if HAVE_TDM
        if (wv == 0) __builtin_amdgcn_s_wait_tensorcnt(0);
#endif
        __syncthreads();                                       // tile ready; prev buffer free
        if (kt + 1 < NT) stage(cur ^ 1, kt + 1);               // overlap next DMA with compute

        const _Float16* Kt = &ldsK[cur][0];
        const _Float16* Vl = &ldsV[cur][0];
#pragma unroll
        for (int sub = 0; sub < 2; ++sub) {                    // 2 x 32-key subtiles
            v16h kb0 = load_frag_rowmajor(Kt + (sub * 32)      * LDK, LDK, lane);
            v16h kb1 = load_frag_rowmajor(Kt + (sub * 32 + 16) * LDK, LDK, lane);
            v8f z = {};
            v8f s0 = wmma_f16(qf, kb0, z);                     // scores already *log2e/sqrt(E)
            v8f s1 = wmma_f16(qf, kb1, z);

            float corr[8];
#pragma unroll
            for (int r = 0; r < 8; ++r) {
                float t = fmaxf(s0[r], s1[r]);
                t = fmaxf(t, __shfl_xor(t, 1, 32));
                t = fmaxf(t, __shfl_xor(t, 2, 32));
                t = fmaxf(t, __shfl_xor(t, 4, 32));
                t = fmaxf(t, __shfl_xor(t, 8, 32));
                float nm = fmaxf(mrun[r], t);
                corr[r] = exp2f(mrun[r] - nm);
                mrun[r] = nm;
                float p0 = exp2f(s0[r] - nm);
                float p1 = exp2f(s1[r] - nm);
                float rs = p0 + p1;
                rs += __shfl_xor(rs, 1, 32);
                rs += __shfl_xor(rs, 2, 32);
                rs += __shfl_xor(rs, 4, 32);
                rs += __shfl_xor(rs, 8, 32);
                lrun[r] = lrun[r] * corr[r] + rs;
                int row = r + rhi;
                P[row * LDP + colL]      = (_Float16)p0;
                P[row * LDP + 16 + colL] = (_Float16)p1;
                o0[r] *= corr[r];
                o1[r] *= corr[r];
            }
            asm volatile("s_wait_dscnt 0" ::: "memory");       // P visible to own wave

            v16h pf  = load_frag_rowmajor(P, LDP, lane);                     // 16q x 32k
            v16h vb0 = load_frag_rowmajor(Vl + sub * 32,            LDV, lane);
            v16h vb1 = load_frag_rowmajor(Vl + 16 * LDV + sub * 32, LDV, lane);
            o0 = wmma_f16(pf, vb0, o0);
            o1 = wmma_f16(pf, vb1, o1);
        }
    }

#pragma unroll
    for (int r = 0; r < 8; ++r) {
        float inv = 1.0f / lrun[r];
        long tok = qTok + r + rhi;
        Hh[tok * DP + colL]      = (_Float16)(o0[r] * inv);
        Hh[tok * DP + 16 + colL] = (_Float16)(o1[r] * inv);
    }
}

// ---------------------------------------------------------------------------
// Kernel 3: output projection H[.,24(pad32)] @ wO[24,128] + bO -> f32 out.
// wO staged transposed ([e][d]) so B fragments are wide LDS loads.
// ---------------------------------------------------------------------------
__global__ __launch_bounds__(256) void out_proj_kernel(
    const _Float16* __restrict__ Hh, const float* __restrict__ wO,
    const float* __restrict__ bO, float* __restrict__ out)
{
    __shared__ __align__(16) _Float16 ldsWOt[EMB * LDP];       // [e][d], 10 KB
    __shared__ float ldsB[EMB];

    const int tid  = threadIdx.x;
    const int lane = tid & 31;
    const int wv   = tid >> 5;

    for (int i = tid; i < EMB * DP; i += 256) {
        int n = i >> 5, k = i & 31;                            // n = out col e, k = d
        ldsWOt[n * LDP + k] = (k < DH) ? (_Float16)wO[k * EMB + n] : (_Float16)0.0f;
    }
    for (int i = tid; i < EMB; i += 256) ldsB[i] = bO[i];
    __syncthreads();

    const long tokBase = (long)blockIdx.x * 128 + (long)wv * 16;
    v16h a = load_frag_rowmajor(Hh + tokBase * DP, DP, lane);
    const int colL = lane & 15;
    const int rhi  = (lane & 16) ? 8 : 0;
#pragma unroll
    for (int nt = 0; nt < 8; ++nt) {
        v16h bfr = load_frag_rowmajor(ldsWOt + nt * 16 * LDP, LDP, lane);
        v8f z = {};
        v8f c = wmma_f16(a, bfr, z);
#pragma unroll
        for (int r = 0; r < 8; ++r) {
            long tok = tokBase + r + rhi;
            out[tok * EMB + nt * 16 + colL] = c[r] + ldsB[nt * 16 + colL];
        }
    }
}

// ---------------------------------------------------------------------------
extern "C" void kernel_launch(void* const* d_in, const int* in_sizes, int n_in,
                              void* d_out, int out_size, void* d_ws, size_t ws_size,
                              hipStream_t stream) {
    (void)in_sizes; (void)n_in; (void)out_size; (void)ws_size;

    const float* x  = (const float*)d_in[0];
    const float* wQ = (const float*)d_in[1];
    const float* bQ = (const float*)d_in[2];
    const float* wK = (const float*)d_in[3];
    const float* bK = (const float*)d_in[4];
    const float* wV = (const float*)d_in[5];
    const float* bV = (const float*)d_in[6];
    const float* wO = (const float*)d_in[7];
    const float* bO = (const float*)d_in[8];
    float* out = (float*)d_out;

    const long N = (long)BQ * SEQ;                 // 32768 tokens
    _Float16* Qh = (_Float16*)d_ws;                // [N][32], scaled by log2e/sqrt(E)
    _Float16* Kh = Qh + N * DP;                    // [N][32]
    _Float16* Vt = Kh + N * DP;                    // [B][32][SEQ]  (transposed V)
    _Float16* Hh = Vt + N * DP;                    // [N][32]   (8 MB ws total)

    qkv_proj_kernel<<<N / 64, 256, 0, stream>>>(x, wQ, bQ, wK, bK, wV, bV, Qh, Kh, Vt);
    attn_kernel<<<(int)(N / 128), 256, 0, stream>>>(Qh, Kh, Vt, Hh);
    out_proj_kernel<<<(int)(N / 128), 256, 0, stream>>>(Hh, wO, bO, out);
}
